// RecurrentCNNModel_55344948576740
// MI455X (gfx1250) — matrix operand order repaired and checked by reading, not compiled, not run on hardware
//
#include <hip/hip_runtime.h>
#include <hip/hip_bf16.h>

// ---------------- problem constants ----------------
constexpr int V  = 50000, E = 300, C = 100, H = 100, NC = 4;
constexpr int NB = 128, L = 1024;
constexpr int NL = NB * L;           // 131072 rows
constexpr int FS = 544;              // feat row stride: [x:0..300 | 0pad:300..320 | cl:320..420 | cr:420..520 | 0pad:520..544]
constexpr int MT = NL / 16;          // 8192 M-tiles

typedef __bf16 bf;
typedef __attribute__((ext_vector_type(16))) __bf16 v16bf;
typedef __attribute__((ext_vector_type(8)))  __bf16 v8bf;
typedef __attribute__((ext_vector_type(8)))  float  v8f;

// ---------------- workspace layout (bytes) ----------------
constexpr size_t FEAT_OFF = 0;                                  // bf16 [NL][544]
constexpr size_t FEAT_SZ  = (size_t)NL * FS * 2;                // 142,606,336
constexpr size_t SL_OFF   = FEAT_OFF + FEAT_SZ;                 // f32 [NL][100]
constexpr size_t SL_SZ    = (size_t)NL * C * 4;
constexpr size_t SR_OFF   = SL_OFF + SL_SZ;
constexpr size_t WSL_OFF  = SR_OFF + SL_SZ;                     // bf16 [112][320]
constexpr size_t WSL_SZ   = 112 * 320 * 2;
constexpr size_t WSR_OFF  = WSL_OFF + WSL_SZ;
constexpr size_t WL_OFF   = WSR_OFF + WSL_SZ;                   // bf16 [112][128]
constexpr size_t WL_SZ    = 112 * 128 * 2;
constexpr size_t WR_OFF   = WL_OFF + WL_SZ;
constexpr size_t W2_OFF   = WR_OFF + WL_SZ;                     // bf16 [112][544]
constexpr size_t W2_SZ    = 112 * 544 * 2;
constexpr size_t Y3_OFF   = W2_OFF + W2_SZ;                     // u32 [128][112] (encoded running max)
constexpr size_t Y3_SZ    = 128 * 112 * 4;

// ---------------- helpers ----------------
__device__ __forceinline__ unsigned enc_f32(float f) {
    unsigned u = __float_as_uint(f);
    return (u >> 31) ? ~u : (u | 0x80000000u);   // monotonic total order
}
__device__ __forceinline__ float dec_f32(unsigned e) {
    unsigned u = (e & 0x80000000u) ? (e ^ 0x80000000u) : ~e;
    return __uint_as_float(u);
}

// A operand: per-lane 16 contiguous K values starting at p (K = kstep*32 + 16*half + e)
__device__ __forceinline__ v16bf load_a(const bf* p) {
    v8bf lo = *(const v8bf*)p;
    v8bf hi = *(const v8bf*)(p + 8);
    return __builtin_shufflevector(lo, hi, 0,1,2,3,4,5,6,7,8,9,10,11,12,13,14,15);
}
// B operand: per-lane 16 contiguous K values starting at p
__device__ __forceinline__ v16bf load_b(const bf* p) {
    v8bf lo = *(const v8bf*)p;
    v8bf hi = *(const v8bf*)(p + 8);
    return __builtin_shufflevector(lo, hi, 0,1,2,3,4,5,6,7,8,9,10,11,12,13,14,15);
}
__device__ __forceinline__ v8f wmma_bf16(v16bf a, v16bf b, v8f c) {
    return __builtin_amdgcn_wmma_f32_16x16x32_bf16(false, a, false, b, (short)0, c, false, false);
}

// ---------------- K0: weight repack (f32 -> padded bf16) + y3max init ----------------
__global__ void prep_kernel(const float* Wl_w, const float* Wr_w,
                            const float* Wsl_w, const float* Wsr_w, const float* W2_w,
                            bf* Wl_p, bf* Wr_p, bf* Wsl_p, bf* Wsr_p, bf* W2_p,
                            unsigned* y3max) {
    int t = blockIdx.x * blockDim.x + threadIdx.x;
    if (t < 14336) {                                   // Wl_p [112][128]
        int n = t / 128, k = t % 128;
        Wl_p[t] = (n < C && k < C) ? (bf)Wl_w[n * C + k] : (bf)0.0f;  return;
    }
    t -= 14336;
    if (t < 14336) {                                   // Wr_p
        int n = t / 128, k = t % 128;
        Wr_p[t] = (n < C && k < C) ? (bf)Wr_w[n * C + k] : (bf)0.0f;  return;
    }
    t -= 14336;
    if (t < 35840) {                                   // Wsl_p [112][320]
        int n = t / 320, k = t % 320;
        Wsl_p[t] = (n < C && k < E) ? (bf)Wsl_w[n * E + k] : (bf)0.0f;  return;
    }
    t -= 35840;
    if (t < 35840) {                                   // Wsr_p
        int n = t / 320, k = t % 320;
        Wsr_p[t] = (n < C && k < E) ? (bf)Wsr_w[n * E + k] : (bf)0.0f;  return;
    }
    t -= 35840;
    if (t < 60928) {                                   // W2_p [112][544], feat-order remap
        int n = t / 544, k = t % 544;
        float v = 0.0f;
        if (n < H) {
            if (k < 300)                 v = W2_w[n * 500 + 100 + k];          // x block
            else if (k >= 320 && k < 420) v = W2_w[n * 500 + (k - 320)];       // cl block
            else if (k >= 420 && k < 520) v = W2_w[n * 500 + 400 + (k - 420)]; // cr block
        }
        W2_p[t] = (bf)v;  return;
    }
    t -= 60928;
    if (t < 14336) y3max[t] = enc_f32(-2.0f);          // tanh >= -1
}

// ---------------- K1: embedding gather -> feat x-region (bf16) + zero pads ----------------
__global__ void gather_kernel(const int* __restrict__ text, const float* __restrict__ emb,
                              bf* __restrict__ feat) {
    int row  = (blockIdx.x * blockDim.x + threadIdx.x) >> 5;   // one wave per (n,l)
    int lane = threadIdx.x & 31;
    long long tok = (long long)text[row];
    const float* e = emb + tok * E;
    bf* fr = feat + (size_t)row * FS;
    for (int c = lane; c < E; c += 32)         fr[c] = (bf)e[c];
    for (int c = 300 + lane; c < 320; c += 32) fr[c] = (bf)0.0f;
    for (int c = 520 + lane; c < 544; c += 32) fr[c] = (bf)0.0f;
}

// ---------------- K2: sl / sr = x @ Wsl^T + b  (WMMA bf16, f32 acc) ----------------
__global__ void gemm_s_kernel(const bf* __restrict__ feat,
                              const bf* __restrict__ Wsl_p, const float* __restrict__ Wsl_b,
                              const bf* __restrict__ Wsr_p, const float* __restrict__ Wsr_b,
                              float* __restrict__ sl, float* __restrict__ sr) {
    const int lane = threadIdx.x, half = lane >> 4, mrow = lane & 15;
    const bool isR = (blockIdx.y != 0);
    const bf*    Wp  = isR ? Wsr_p : Wsl_p;
    const float* Wb  = isR ? Wsr_b : Wsl_b;
    float*       out = isR ? sr : sl;
    const size_t row0 = (size_t)blockIdx.x * 16;

    v8f acc[7] = {};
    const bf* Abase = feat + (row0 + mrow) * FS + half * 16;   // x region at col 0
#pragma unroll
    for (int ks = 0; ks < 10; ++ks) {                          // K = 320
        v16bf a = load_a(Abase + ks * 32);
#pragma unroll
        for (int nt = 0; nt < 7; ++nt) {
            const bf* pb = Wp + (size_t)(nt * 16 + mrow) * 320 + ks * 32 + half * 16;
            acc[nt] = wmma_bf16(a, load_b(pb), acc[nt]);
        }
    }
#pragma unroll
    for (int nt = 0; nt < 7; ++nt) {
        int col = nt * 16 + mrow;
        if (col < C) {
            float bias = Wb[col];
#pragma unroll
            for (int r = 0; r < 8; ++r) {
                size_t grow = row0 + r + half * 8;
                out[grow * C + col] = acc[nt][r] + bias;
            }
        }
    }
}

// ---------------- K3: recurrent scans (left & right concurrently via gridDim.y) ----------------
// 7 waves (224 thr) per block; block owns 16 batch rows; state double-buffered in LDS.
// Per-wave B tiles (4 k-steps of its N-tile) are loop-invariant -> preloaded into registers,
// so the 1023-step critical path is only: ds_load A -> 4 chained WMMA -> epilogue -> barrier.
__global__ void __launch_bounds__(224) scan_kernel(
        const bf* __restrict__ Wl_p, const float* __restrict__ Wl_b,
        const bf* __restrict__ Wr_p, const float* __restrict__ Wr_b,
        const float* __restrict__ sl, const float* __restrict__ sr,
        bf* __restrict__ feat) {
    const int dir = blockIdx.y;                 // 0 = left, 1 = right
    const bf*    Wp    = dir ? Wr_p : Wl_p;
    const float* Wb    = dir ? Wr_b : Wl_b;
    const float* sbuf  = dir ? sr : sl;
    const int   colOff = dir ? 420 : 320;
    const int   b0     = blockIdx.x * 16;

    __shared__ bf state[2][16 * 128];
    for (int i = threadIdx.x; i < 2 * 16 * 128; i += blockDim.x)
        ((bf*)state)[i] = (bf)0.0f;
    {   // boundary column (cl[0] / cr[L-1]) = 0
        int lB = dir ? (L - 1) : 0;
        for (int i = threadIdx.x; i < 16 * C; i += blockDim.x) {
            int m = i / C, c = i % C;
            feat[((size_t)(b0 + m) * L + lB) * FS + colOff + c] = (bf)0.0f;
        }
    }
    __syncthreads();

    const int w = threadIdx.x >> 5;             // wave -> N-tile (0..6)
    const int lane = threadIdx.x & 31, half = lane >> 4, mrow = lane & 15;
    const int col = w * 16 + mrow;
    const float bias = (col < C) ? Wb[col] : 0.0f;

    // preload loop-invariant B tiles (weights) into registers
    v16bf bmat[4];
#pragma unroll
    for (int ks = 0; ks < 4; ++ks)
        bmat[ks] = load_b(Wp + (size_t)(w * 16 + mrow) * 128 + ks * 32 + half * 16);

    int p = 0;
    for (int step = 1; step < L; ++step) {
        const int l    = dir ? (L - 1 - step) : step;
        const int sIdx = dir ? (l + 1) : (l - 1);
        v8f acc = {};
#pragma unroll
        for (int ks = 0; ks < 4; ++ks) {        // K = 128 (state cols 100..127 stay zero)
            v16bf a = load_a(&state[p][mrow * 128 + ks * 32 + half * 16]);
            acc = wmma_bf16(a, bmat[ks], acc);
        }
        if (col < C) {
#pragma unroll
            for (int r = 0; r < 8; ++r) {
                int m = r + half * 8;
                size_t grow = (size_t)(b0 + m);
                float f = acc[r] + bias + sbuf[(grow * L + sIdx) * C + col];
                f = f > 0.0f ? f : 0.0f;
                bf h = (bf)f;
                feat[(grow * L + l) * FS + colOff + col] = h;
                state[p ^ 1][m * 128 + col] = h;
            }
        }
        __syncthreads();
        p ^= 1;
    }
}

// ---------------- K4: y2 = tanh(feat @ W2^T + b2); running max over time (atomic) ----------------
__global__ void __launch_bounds__(224) gemm2_kernel(
        const bf* __restrict__ feat, const bf* __restrict__ W2_p,
        const float* __restrict__ W2_b, unsigned* __restrict__ y3max) {
    const int w = threadIdx.x >> 5;             // wave -> N-tile (0..6)
    const int lane = threadIdx.x & 31, half = lane >> 4, mrow = lane & 15;
    const size_t row0 = (size_t)blockIdx.x * 16;
    const int n = (int)(row0 >> 10);            // 16 consecutive l in one batch row

    v8f acc = {};
    const bf* Abase = feat + (row0 + mrow) * FS + half * 16;
#pragma unroll
    for (int ks = 0; ks < 17; ++ks) {           // K = 544
        v16bf a = load_a(Abase + ks * 32);
        const bf* pb = W2_p + (size_t)(w * 16 + mrow) * FS + ks * 32 + half * 16;
        acc = wmma_bf16(a, load_b(pb), acc);
    }
    const int col = w * 16 + mrow;
    float m = -2.0f;
    if (col < H) {
        float bias = W2_b[col];
#pragma unroll
        for (int r = 0; r < 8; ++r) {
            float v = tanhf(acc[r] + bias);
            m = fmaxf(m, v);
        }
    }
    m = fmaxf(m, __shfl_xor(m, 16, 32));        // combine the two row-halves
    if (half == 0 && col < H)
        atomicMax(&y3max[n * 112 + col], enc_f32(m));
}

// ---------------- K5: head  out = y3 @ W4^T + b4 ----------------
__global__ void final_kernel(const unsigned* __restrict__ y3max,
                             const float* __restrict__ W4_w, const float* __restrict__ W4_b,
                             float* __restrict__ out) {
    int t = blockIdx.x * blockDim.x + threadIdx.x;
    if (t >= NB * NC) return;
    int n = t >> 2, cls = t & 3;
    float sum = W4_b[cls];
    for (int h = 0; h < H; ++h)
        sum += dec_f32(y3max[n * 112 + h]) * W4_w[cls * H + h];
    out[n * NC + cls] = sum;
}

// ---------------- launch ----------------
extern "C" void kernel_launch(void* const* d_in, const int* in_sizes, int n_in,
                              void* d_out, int out_size, void* d_ws, size_t ws_size,
                              hipStream_t stream) {
    const int*   text  = (const int*)  d_in[0];
    const float* emb   = (const float*)d_in[1];
    const float* Wl_w  = (const float*)d_in[2];
    const float* Wl_b  = (const float*)d_in[3];
    const float* Wr_w  = (const float*)d_in[4];
    const float* Wr_b  = (const float*)d_in[5];
    const float* Wsl_w = (const float*)d_in[6];
    const float* Wsl_b = (const float*)d_in[7];
    const float* Wsr_w = (const float*)d_in[8];
    const float* Wsr_b = (const float*)d_in[9];
    const float* W2_w  = (const float*)d_in[10];
    const float* W2_b  = (const float*)d_in[11];
    const float* W4_w  = (const float*)d_in[12];
    const float* W4_b  = (const float*)d_in[13];

    char* ws = (char*)d_ws;
    bf*       feat  = (bf*)(ws + FEAT_OFF);
    float*    sl    = (float*)(ws + SL_OFF);
    float*    sr    = (float*)(ws + SR_OFF);
    bf*       Wsl_p = (bf*)(ws + WSL_OFF);
    bf*       Wsr_p = (bf*)(ws + WSR_OFF);
    bf*       Wl_p  = (bf*)(ws + WL_OFF);
    bf*       Wr_p  = (bf*)(ws + WR_OFF);
    bf*       W2_p  = (bf*)(ws + W2_OFF);
    unsigned* y3max = (unsigned*)(ws + Y3_OFF);
    float*    out   = (float*)d_out;

    // K0: weight repack + y3max init (176k work items)
    prep_kernel<<<686, 256, 0, stream>>>(Wl_w, Wr_w, Wsl_w, Wsr_w, W2_w,
                                         Wl_p, Wr_p, Wsl_p, Wsr_p, W2_p, y3max);
    // K1: gather (one wave per row, 8 rows per block)
    gather_kernel<<<NL / 8, 256, 0, stream>>>(text, emb, feat);
    // K2: sl & sr projections
    gemm_s_kernel<<<dim3(MT, 2), 32, 0, stream>>>(feat, Wsl_p, Wsl_b, Wsr_p, Wsr_b, sl, sr);
    // K3: left & right recurrences, concurrent
    scan_kernel<<<dim3(NB / 16, 2), 224, 0, stream>>>(Wl_p, Wl_b, Wr_p, Wr_b, sl, sr, feat);
    // K4: big fused GEMM + tanh + time max-pool
    gemm2_kernel<<<MT, 224, 0, stream>>>(feat, W2_p, W2_b, y3max);
    // K5: classifier head
    final_kernel<<<2, 256, 0, stream>>>(y3max, W4_w, W4_b, out);
}
